// Attention_LSTM_79379585565596
// MI455X (gfx1250) — compile-verified, matrix operand order checked by reading, not run on hardware
//
#include <hip/hip_runtime.h>

#define BB 64
#define SS 1024
#define DD 512
#define GG 2048   // 4*D
#define OUTN 2
#define NBLK 32   // persistent scan blocks

typedef __attribute__((ext_vector_type(16))) __bf16 v16bf;
typedef __attribute__((ext_vector_type(8)))  float  v8f;
typedef unsigned short u16;
typedef unsigned int   u32;

// fp32 -> bf16 bits, round-to-nearest-even
__device__ __forceinline__ u16 f2bf(float f) {
  u32 u = __float_as_uint(f);
  u32 r = u + 0x7FFFu + ((u >> 16) & 1u);
  return (u16)(r >> 16);
}

// A-operand fragment from row-major storage (stride ld, multiple of 8):
// per CDNA5 16-bit mapping each lane's 16 elements are two contiguous 16B runs.
__device__ __forceinline__ v16bf load_frag_rm(const u16* tile, int ld, int lane) {
  const int row  = lane & 15;
  const int half = lane >> 4;
  const u16* p = tile + row * ld + half * 8;
  union { v16bf v; uint4 q[2]; } f;
  f.q[0] = *(const uint4*)p;         // pairs 0..3
  f.q[1] = *(const uint4*)(p + 16);  // pairs 4..7
  return f.v;
}

// B-operand fragment from swizzled (fragment-native) tile: lane*32B contiguous.
__device__ __forceinline__ v16bf load_frag_swz(const u16* tileBase, int lane) {
  const u16* p = tileBase + lane * 16;
  union { v16bf v; uint4 q[2]; } f;
  f.q[0] = ((const uint4*)p)[0];
  f.q[1] = ((const uint4*)p)[1];
  return f.v;
}

__device__ __forceinline__ v8f wmma_bf16(v16bf a, v16bf b, v8f c) {
  return __builtin_amdgcn_wmma_f32_16x16x32_bf16(false, a, false, b, (short)0, c,
                                                 false, false);
}

// CDNA5 async global->LDS copy (16B per lane), tracked by ASYNCcnt.
__device__ __forceinline__ void async_b128(u16* ldsDst, const u16* gsrc) {
  u32 off = (u32)(size_t)ldsDst;  // low 32 bits of flat LDS address = LDS offset
  asm volatile("global_load_async_to_lds_b128 %0, %1, off"
               :: "v"(off), "v"(gsrc) : "memory");
}
template <int N>
__device__ __forceinline__ void wait_async() {
  asm volatile("s_wait_asynccnt %0" :: "i"(N) : "memory");
}

// ---------------------------------------------------------------------------
// Big WMMA GEMM: C[M,N] = A[M,K] * W_swz[N,K]^T + bias[N]
// A fp32 (converted on the fly) or bf16, reg double-buffered through LDS;
// B (pre-swizzled tiles) staged via async global->LDS, double-buffered.
// 256 threads (8 waves), 128x128 tile, K-chunk 32.
// ---------------------------------------------------------------------------
struct Stage { union { u16 h[16]; uint4 q[2]; }; };

template <bool A_F32, bool STORE_BF16>
__global__ __launch_bounds__(256) void gemm_wmma(
    const void* __restrict__ Ap, const u16* __restrict__ W,
    const float* __restrict__ bias, float* __restrict__ Cf,
    u16* __restrict__ Cb, int M, int N, int K) {
  const int LD = 40;  // A pitch: 32 + 8 pad; 80B rows keep 16B alignment
  __shared__ u16 As[128 * LD];
  __shared__ u16 Bs[2][8 * 512];  // 8 swizzled 16x32 tiles per buffer

  const int tid  = threadIdx.x;
  const int lane = tid & 31;
  const int wid  = tid >> 5;
  const int wm0  = (wid & 3) * 32;
  const int wn16 = (wid >> 2) * 4;  // wave's first n-tile (of 8)
  const long bm = (long)blockIdx.x * 128;
  const long bn16 = (long)blockIdx.y * 8;  // n-tile base
  const int kt = K / 32;

  v8f zero = {};
  v8f acc[2][4];
#pragma unroll
  for (int i = 0; i < 2; ++i)
#pragma unroll
    for (int j = 0; j < 4; ++j) acc[i][j] = zero;

  const int r    = tid >> 1;
  const int cseg = (tid & 1) * 16;

  auto loadA = [&](int kc, Stage& t) {
    if (A_F32) {
      const float* A = (const float*)Ap + (bm + r) * (long)K + kc + cseg;
#pragma unroll
      for (int v = 0; v < 4; ++v) {
        float4 x = ((const float4*)A)[v];
        t.h[4 * v + 0] = f2bf(x.x); t.h[4 * v + 1] = f2bf(x.y);
        t.h[4 * v + 2] = f2bf(x.z); t.h[4 * v + 3] = f2bf(x.w);
      }
    } else {
      const u16* A = (const u16*)Ap + (bm + r) * (long)K + kc + cseg;
      t.q[0] = ((const uint4*)A)[0];
      t.q[1] = ((const uint4*)A)[1];
    }
  };
  // async-copy 8 swizzled B tiles (8KB) for k-chunk index kci into Bs[buf]
  auto issueB = [&](int kci, int buf) {
    const int ti = tid >> 5;   // tile 0..7
    const int sl = tid & 31;
    const u16* src = W + (((long)(bn16 + ti) * kt + kci) * 512) + sl * 16;
    u16* dst = &Bs[buf][ti * 512 + sl * 16];
    async_b128(dst, src);
    async_b128(dst + 8, src + 8);
  };

  Stage ca, na;
  loadA(0, ca);
  issueB(0, 0);

  for (int it = 0; it < kt; ++it) {
    const int kc = it * 32;
    const int p  = it & 1;
    *(uint4*)&As[r * LD + cseg]     = ca.q[0];
    *(uint4*)&As[r * LD + cseg + 8] = ca.q[1];

    const bool more = (it + 1) < kt;
    if (more) {
      loadA(kc + 32, na);
      issueB(it + 1, p ^ 1);
      wait_async<2>();       // current chunk done; next chunk's 2 ops in flight
    } else {
      wait_async<0>();
    }
    __syncthreads();

    v16bf af[2], bf[4];
#pragma unroll
    for (int i = 0; i < 2; ++i) af[i] = load_frag_rm(&As[(wm0 + 16 * i) * LD], LD, lane);
#pragma unroll
    for (int j = 0; j < 4; ++j) bf[j] = load_frag_swz(&Bs[p][(wn16 + j) * 512], lane);
#pragma unroll
    for (int i = 0; i < 2; ++i)
#pragma unroll
      for (int j = 0; j < 4; ++j) acc[i][j] = wmma_bf16(af[i], bf[j], acc[i][j]);
    __syncthreads();
    if (more) ca = na;
  }

  const int nl = lane & 15, half = lane >> 4;
#pragma unroll
  for (int i = 0; i < 2; ++i)
#pragma unroll
    for (int j = 0; j < 4; ++j) {
      const long row0 = bm + wm0 + 16 * i;
      const long col0 = (bn16 + wn16 + j) * 16;
#pragma unroll
      for (int rr = 0; rr < 8; ++rr) {
        long m = row0 + 8 * half + rr;
        long n = col0 + nl;
        float v = acc[i][j][rr] + bias[n];
        Cf[m * (long)N + n] = v;
        if (STORE_BF16) Cb[m * (long)N + n] = f2bf(v);
      }
    }
}

// ---------------------------------------------------------------------------
// Persistent LSTM scan: 32 blocks, all S steps in one launch.
// W_hh slice LDS-resident as 64 swizzled 1KB tiles -> each B fragment is
// 2x ds_load_b128, no address math. Split-K (4 WMMA chains/wave) for hazards.
// Dynamic LDS: 64*1024 (weights) + 4*64*16*4 (gate xchg) = 81,920 B.
// ---------------------------------------------------------------------------
__global__ __launch_bounds__(256) void lstm_scan(
    const u16* __restrict__ Whh, const float* __restrict__ xg,
    u16* __restrict__ h0buf, u16* __restrict__ h1buf,
    float* __restrict__ c, float* __restrict__ hF, int* __restrict__ arrive) {
  extern __shared__ char smem[];
  u16*   wl = (u16*)smem;                    // [4 gates][16 k-tiles][512]
  float* gl = (float*)(smem + 64 * 1024);    // [4][64][16]

  const int tid = threadIdx.x, lane = tid & 31, wid = tid >> 5;
  const int ns16 = blockIdx.x;            // d-slice tile index (16-wide)
  const int ns = ns16 * 16;
  const int m0 = (wid >> 1) * 16;
  const int g0 = (2 * wid) & 3, g1 = (2 * wid + 1) & 3;

  // --- preload this block's 64 swizzled W_hh tiles into LDS (once) ---
  for (int ti = wid; ti < 64; ti += 8) {
    const int g = ti >> 4, ktile = ti & 15;
    const u16* src = Whh + ((long)(g * 32 + ns16) * 16 + ktile) * 512 + lane * 16;
    u16* dst = wl + ti * 512 + lane * 16;
    ((uint4*)dst)[0] = ((const uint4*)src)[0];
    ((uint4*)dst)[1] = ((const uint4*)src)[1];
  }
  __syncthreads();

  const int nl = lane & 15, half = lane >> 4;

  for (int s = 0; s < SS; ++s) {
    const u16* hin  = (s & 1) ? h1buf : h0buf;
    u16*       hout = (s & 1) ? h0buf : h1buf;

    v8f a0l = {}, a0h = {}, a1l = {}, a1h = {};
#pragma unroll
    for (int kk = 0; kk < 8; ++kk) {
      {
        const int ki = kk * 2;
        v16bf a  = load_frag_rm(hin + m0 * 512 + ki * 32, 512, lane);
        v16bf b0 = load_frag_swz(wl + (g0 * 16 + ki) * 512, lane);
        v16bf b1 = load_frag_swz(wl + (g1 * 16 + ki) * 512, lane);
        a0l = wmma_bf16(a, b0, a0l);
        a1l = wmma_bf16(a, b1, a1l);
      }
      {
        const int ki = kk * 2 + 1;
        v16bf a  = load_frag_rm(hin + m0 * 512 + ki * 32, 512, lane);
        v16bf b0 = load_frag_swz(wl + (g0 * 16 + ki) * 512, lane);
        v16bf b1 = load_frag_swz(wl + (g1 * 16 + ki) * 512, lane);
        a0h = wmma_bf16(a, b0, a0h);
        a1h = wmma_bf16(a, b1, a1h);
      }
    }
    v8f acc0 = a0l + a0h;
    v8f acc1 = a1l + a1h;

#pragma unroll
    for (int rr = 0; rr < 8; ++rr) {
      gl[(g0 * 64 + m0 + 8 * half + rr) * 16 + nl] = acc0[rr];
      gl[(g1 * 64 + m0 + 8 * half + rr) * 16 + nl] = acc1[rr];
    }
    __syncthreads();

    for (int e = tid; e < 1024; e += 256) {
      int b = e >> 4, dl = e & 15, d = ns + dl;
      long xrow = ((long)b * SS + s) * GG;
      float ip = gl[(0 * 64 + b) * 16 + dl] + xg[xrow + d];
      float fp = gl[(1 * 64 + b) * 16 + dl] + xg[xrow + 512 + d];
      float gp = gl[(2 * 64 + b) * 16 + dl] + xg[xrow + 1024 + d];
      float op = gl[(3 * 64 + b) * 16 + dl] + xg[xrow + 1536 + d];
      if (s + 1 < SS) {  // prefetch next step's xg slice (global_prefetch_b8)
        __builtin_prefetch(&xg[xrow + GG + d], 0, 0);
        __builtin_prefetch(&xg[xrow + GG + 512 + d], 0, 0);
        __builtin_prefetch(&xg[xrow + GG + 1024 + d], 0, 0);
        __builtin_prefetch(&xg[xrow + GG + 1536 + d], 0, 0);
      }
      float iv = 1.f / (1.f + __expf(-ip));
      float fv = 1.f / (1.f + __expf(-fp));
      float gv = tanhf(gp);
      float ov = 1.f / (1.f + __expf(-op));
      float cn = fv * c[b * 512 + d] + iv * gv;
      c[b * 512 + d] = cn;
      float hn = ov * tanhf(cn);
      hF[b * 512 + d] = hn;
      hout[b * 512 + d] = f2bf(hn);
    }

    // device-wide step barrier: release h writes, arrive, spin, acquire
    __threadfence();
    __syncthreads();
    if (tid == 0) {
      atomicAdd(&arrive[s], 1);
      while (atomicAdd(&arrive[s], 0) < NBLK) __builtin_amdgcn_s_sleep(1);
    }
    __syncthreads();
    __threadfence();
  }
}

// --------------------------- attention + head ------------------------------
__global__ __launch_bounds__(256) void attn_logits(
    const float* __restrict__ hF, const float* __restrict__ feat,
    float* __restrict__ logits) {
  int gw   = (blockIdx.x * 256 + threadIdx.x) >> 5;  // one wave per (b,s)
  int lane = threadIdx.x & 31;
  int b = gw >> 10, s = gw & (SS - 1);
  const float* f = feat + ((long)b * SS + s) * DD;
  const float* h = hF + (long)b * DD;
  float acc = 0.f;
  for (int k = lane; k < DD; k += 32) acc = fmaf(f[k], h[k], acc);
#pragma unroll
  for (int off = 16; off; off >>= 1) acc += __shfl_xor(acc, off, 32);
  if (lane == 0) logits[(long)b * SS + s] = acc;
}

__global__ __launch_bounds__(256) void softmax_rows(
    const float* __restrict__ logits, float* __restrict__ w) {
  __shared__ float red[256];
  int b = blockIdx.x, tid = threadIdx.x;
  const float* L = logits + (long)b * SS;
  float m = -3.4e38f;
  for (int s = tid; s < SS; s += 256) m = fmaxf(m, L[s]);
  red[tid] = m; __syncthreads();
  for (int o = 128; o; o >>= 1) { if (tid < o) red[tid] = fmaxf(red[tid], red[tid + o]); __syncthreads(); }
  m = red[0]; __syncthreads();
  float sum = 0.f;
  float* W = w + (long)b * SS;
  for (int s = tid; s < SS; s += 256) { float e = __expf(L[s] - m); W[s] = e; sum += e; }
  red[tid] = sum; __syncthreads();
  for (int o = 128; o; o >>= 1) { if (tid < o) red[tid] += red[tid + o]; __syncthreads(); }
  float inv = 1.f / red[0];
  for (int s = tid; s < SS; s += 256) W[s] *= inv;
}

// combined = weights @ feat, two deterministic stages (no float atomics)
__global__ __launch_bounds__(256) void attn_combine_part(
    const float* __restrict__ w, const float* __restrict__ feat,
    float* __restrict__ part) {
  int b = blockIdx.x, ch = blockIdx.y, tid = threadIdx.x;  // 16 s-chunks of 64
  const float* W = w + (long)b * SS + ch * 64;
  const float* F = feat + ((long)b * SS + ch * 64) * DD;
  for (int d = tid; d < DD; d += 256) {
    float a = 0.f;
#pragma unroll 4
    for (int s = 0; s < 64; ++s) a = fmaf(W[s], F[(long)s * DD + d], a);
    part[((long)b * 16 + ch) * DD + d] = a;
  }
}

__global__ __launch_bounds__(256) void attn_combine_red(
    const float* __restrict__ part, float* __restrict__ comb) {
  int b = blockIdx.x, tid = threadIdx.x;
  for (int d = tid; d < DD; d += 256) {
    float a = 0.f;
#pragma unroll
    for (int ch = 0; ch < 16; ++ch) a += part[((long)b * 16 + ch) * DD + d];
    comb[b * DD + d] = a;
  }
}

__global__ void fc2_kernel(const float* __restrict__ comb,
                           const float* __restrict__ W2,
                           const float* __restrict__ b2, float* __restrict__ out) {
  int t = blockIdx.x * blockDim.x + threadIdx.x;
  if (t >= BB * OUTN) return;
  int b = t / OUTN, o = t % OUTN;
  float a = b2[o];
  for (int k = 0; k < DD; ++k) a = fmaf(comb[b * DD + k], W2[o * DD + k], a);
  out[t] = a;
}

// ------------------------------ prep ---------------------------------------
// Convert fp32 weights [N][K] to bf16 in fragment-native swizzled tile order:
// tile(nt,kt) is 512 elems at ((nt*(K/32)+kt)*512); element (n,k) sits at
// lane*16+e with lane = (n&15) + 16*((k>>3)&1), e = 2*(4*((k>>4)&1)+((k>>1)&3)) + (k&1).
__global__ void cvt_swz_kernel(const float* __restrict__ x, u16* __restrict__ y,
                               int N, int K) {
  long i = (long)blockIdx.x * 256 + threadIdx.x;
  if (i >= (long)N * K) return;
  int n = (int)(i / K), k = (int)(i % K);
  int nt = n >> 4, nl = n & 15, ktile = k >> 5, kk = k & 31;
  int half = (kk >> 3) & 1, bit = kk & 1;
  int p = 4 * ((kk >> 4) & 1) + ((kk >> 1) & 3);
  int lane = nl + 16 * half, e = 2 * p + bit;
  y[((long)nt * (K / 32) + ktile) * 512 + lane * 16 + e] = f2bf(x[i]);
}

__global__ void prep_bias_state(const float* __restrict__ b_ih,
                                const float* __restrict__ b_hh,
                                float* __restrict__ bias2,
                                const float* __restrict__ h0,
                                const float* __restrict__ c0,
                                u16* __restrict__ hA, float* __restrict__ hF,
                                float* __restrict__ cF, int* __restrict__ arrive) {
  int i = blockIdx.x * 256 + threadIdx.x;
  if (i < GG) bias2[i] = b_ih[i] + b_hh[i];
  if (i < SS) arrive[i] = 0;                      // reset step barrier counters
  if (i < BB * DD) { hA[i] = f2bf(h0[i]); hF[i] = h0[i]; cF[i] = c0[i]; }
}

// ---------------------------------------------------------------------------
extern "C" void kernel_launch(void* const* d_in, const int* in_sizes, int n_in,
                              void* d_out, int out_size, void* d_ws, size_t ws_size,
                              hipStream_t stream) {
  const float* aa_feature = (const float*)d_in[0];
  const float* aa_h0      = (const float*)d_in[1];
  const float* aa_c0      = (const float*)d_in[2];
  const float* W_fc       = (const float*)d_in[3];
  const float* b_fc       = (const float*)d_in[4];
  const float* W_ih       = (const float*)d_in[5];
  const float* W_hh       = (const float*)d_in[6];
  const float* b_ih       = (const float*)d_in[7];
  const float* b_hh       = (const float*)d_in[8];
  const float* W_fc2      = (const float*)d_in[9];
  const float* b_fc2      = (const float*)d_in[10];
  (void)in_sizes; (void)n_in; (void)out_size; (void)ws_size;

  char* ws = (char*)d_ws;
  size_t off = 0;
  auto alloc = [&](size_t bytes) {
    void* p = ws + off;
    off = (off + bytes + 255) & ~(size_t)255;
    return p;
  };
  float* featF = (float*)alloc((size_t)BB * SS * DD * 4);
  u16*   featB = (u16*)  alloc((size_t)BB * SS * DD * 2);
  float* xg    = (float*)alloc((size_t)BB * SS * GG * 4);
  u16*   wfcB  = (u16*)  alloc((size_t)DD * DD * 2);
  u16*   wihB  = (u16*)  alloc((size_t)GG * DD * 2);
  u16*   whhB  = (u16*)  alloc((size_t)GG * DD * 2);
  float* bias2 = (float*)alloc((size_t)GG * 4);
  u16*   hA    = (u16*)  alloc((size_t)BB * DD * 2);
  u16*   hB    = (u16*)  alloc((size_t)BB * DD * 2);
  float* hF    = (float*)alloc((size_t)BB * DD * 4);
  float* cF    = (float*)alloc((size_t)BB * DD * 4);
  float* logit = (float*)alloc((size_t)BB * SS * 4);
  float* wts   = (float*)alloc((size_t)BB * SS * 4);
  float* part  = (float*)alloc((size_t)BB * 16 * DD * 4);
  float* comb  = (float*)alloc((size_t)BB * DD * 4);
  int*   arrv  = (int*)  alloc((size_t)SS * 4);

  cvt_swz_kernel<<<(DD * DD + 255) / 256, 256, 0, stream>>>(W_fc, wfcB, DD, DD);
  cvt_swz_kernel<<<(GG * DD + 255) / 256, 256, 0, stream>>>(W_ih, wihB, GG, DD);
  cvt_swz_kernel<<<(GG * DD + 255) / 256, 256, 0, stream>>>(W_hh, whhB, GG, DD);
  prep_bias_state<<<(BB * DD + 255) / 256, 256, 0, stream>>>(
      b_ih, b_hh, bias2, aa_h0, aa_c0, hA, hF, cF, arrv);

  // feat = aa_feature @ W_fc^T + b_fc   (store fp32 + bf16)
  gemm_wmma<true, true><<<dim3(512, 4), 256, 0, stream>>>(
      aa_feature, wfcB, b_fc, featF, featB, BB * SS, DD, DD);

  // hoisted input-gate GEMM: xg = feat @ W_ih^T + (b_ih + b_hh)
  gemm_wmma<false, false><<<dim3(512, 16), 256, 0, stream>>>(
      featB, wihB, bias2, xg, (u16*)nullptr, BB * SS, GG, DD);

  // persistent recurrence: one launch for all 1024 steps
  lstm_scan<<<NBLK, 256, 64 * 1024 + 4 * 64 * 16 * 4, stream>>>(
      whhB, xg, hA, hB, cF, hF, arrv);

  // attention over projected features + output head
  attn_logits<<<(BB * SS) / 8, 256, 0, stream>>>(hF, featF, logit);
  softmax_rows<<<BB, 256, 0, stream>>>(logit, wts);
  attn_combine_part<<<dim3(BB, 16), 256, 0, stream>>>(wts, featF, part);
  attn_combine_red<<<BB, 256, 0, stream>>>(part, comb);
  fc2_kernel<<<1, 128, 0, stream>>>(comb, W_fc2, b_fc2, (float*)d_out);
}
